// SSDBlock_70403103916652
// MI455X (gfx1250) — matrix operand order repaired and checked by reading, not compiled, not run on hardware
//
#include <hip/hip_runtime.h>
#include <hip/hip_bf16.h>

// Mamba-2 SSD block for MI455X (gfx1250), wave32 + WMMA bf16 (f32 accum).
// WMMA operands staged in LDS K-major (fragments load as ds_load_b128);
// output GEMM uses double-buffered async global->LDS DMA (ASYNCcnt).

#define BATCH   2
#define SEQ     4096
#define DIM     1024
#define HEADS   16
#define DSTATE  64
#define BLOCK_L 64
#define NCHUNK  (SEQ / BLOCK_L)   // 64
#define HEADDIM (DIM / HEADS)     // 64
#define LN_EPS  1e-6f
#define PAD     88                // row pad (bf16): 176B = 16B aligned, 44 banks

typedef __bf16 bf16_t;
typedef bf16_t v16bf __attribute__((ext_vector_type(16)));
typedef bf16_t v8bf  __attribute__((ext_vector_type(8)));
typedef float  v8f   __attribute__((ext_vector_type(8)));

#if defined(__has_builtin)
#if __has_builtin(__builtin_amdgcn_global_load_async_to_lds_b128)
#define HAVE_ASYNC_LDS 1
#endif
#if __has_builtin(__builtin_amdgcn_s_wait_asynccnt)
#define HAVE_WAIT_ASYNC 1
#endif
#endif
#ifndef HAVE_ASYNC_LDS
#define HAVE_ASYNC_LDS 0
#endif
#ifndef HAVE_WAIT_ASYNC
#define HAVE_WAIT_ASYNC 0
#endif

#if HAVE_ASYNC_LDS
// Builtin signature (from hipcc diagnostic): param0 is
// 'int __attribute__((vector_size(16))) __device__ *'  == int4 in AS1 (global);
// param1 is the LDS destination (AS3).
typedef int v4i_vs __attribute__((vector_size(16)));
typedef __attribute__((address_space(1))) v4i_vs* gp128_t;
typedef __attribute__((address_space(3))) v4i_vs* lp128_t;
#endif

// 16-byte global -> LDS copy: async DMA (ASYNCcnt) when available, else sync.
__device__ __forceinline__ void copy16_g2l(void* lds, const void* g) {
#if HAVE_ASYNC_LDS
  gp128_t gp = (gp128_t)(uintptr_t)g;
  lp128_t lp = (lp128_t)(uint32_t)(uintptr_t)lds;  // flat LDS addr truncates to [31:0]
  __builtin_amdgcn_global_load_async_to_lds_b128(gp, lp, 0, 0);
#else
  *(v8bf*)lds = *(const v8bf*)g;
#endif
}

__device__ __forceinline__ void wait_async0() {
#if HAVE_WAIT_ASYNC
  __builtin_amdgcn_s_wait_asynccnt(0);
#elif HAVE_ASYNC_LDS
  asm volatile("s_wait_asynccnt 0x0" ::: "memory");
#endif
}

__device__ __forceinline__ v8f wmma_bf16(v16bf a, v16bf b, v8f c) {
  return __builtin_amdgcn_wmma_f32_16x16x32_bf16(false, a, false, b, (short)0, c,
                                                 false, false);
}

// A-fragment (16x32 bf16): lane reads rows [row][K]: two contiguous 8-elem runs
// at K = kt*32 + (lane>=16)*8 and +16.  -> 2x ds_load_b128
__device__ __forceinline__ v16bf fragA(const bf16_t* rowp, int lane, int kt) {
  const bf16_t* p = rowp + kt * 32 + ((lane >> 4) << 3);
  v8bf lo = *(const v8bf*)p;
  v8bf hi = *(const v8bf*)(p + 16);
  return __builtin_shufflevector(lo, hi, 0, 1, 2, 3, 4, 5, 6, 7, 8, 9, 10, 11,
                                 12, 13, 14, 15);
}
// B-fragment (32x16 bf16) stored K-major ([col][K]): two contiguous 8-elem runs
// at K = kt*32 + (lane>=16)*8 and +16.  -> 2x ds_load_b128
__device__ __forceinline__ v16bf fragB(const bf16_t* rowp, int lane, int kt) {
  const bf16_t* p = rowp + kt * 32 + ((lane >> 4) << 3);
  v8bf lo = *(const v8bf*)p;
  v8bf hi = *(const v8bf*)(p + 16);
  return __builtin_shufflevector(lo, hi, 0, 1, 2, 3, 4, 5, 6, 7, 8, 9, 10, 11,
                                 12, 13, 14, 15);
}

__device__ __forceinline__ float softplus_f(float x) {
  return (x > 20.f) ? x : __logf(1.f + __expf(x));
}

// ---------------- prep: W_out -> bf16 transposed [n][k]; W_dt -> bf16 [h][k]
__global__ void k_prep(const float* __restrict__ wout, const float* __restrict__ wdt,
                       bf16_t* __restrict__ woutT, bf16_t* __restrict__ wdtT) {
  int i = blockIdx.x * blockDim.x + threadIdx.x;
  if (i < DIM * DIM) {
    int k = i >> 10, n = i & (DIM - 1);
    woutT[(size_t)n * DIM + k] = (bf16_t)wout[i];
  }
  if (i < DIM * HEADS) {
    int k = i >> 4, h = i & (HEADS - 1);
    wdtT[(size_t)h * DIM + k] = (bf16_t)wdt[i];
  }
}

// ---------------------------------------------------- dt = softplus(X@Wdt + b)
__global__ void __launch_bounds__(32)
k_dt(const float* __restrict__ X, const bf16_t* __restrict__ WdtT,
     const float* __restrict__ dt_bias, const float* __restrict__ a_log,
     float* __restrict__ dt_out, float* __restrict__ dA_out) {
  const int lane = threadIdx.x;
  const int r0 = blockIdx.x * 16;
  const int row = r0 + (lane & 15);
  const int hcol = lane & 15;
  const bf16_t* wrow = WdtT + (size_t)hcol * DIM;
  const float* xrow = X + (size_t)row * DIM + ((lane >> 4) << 3);
  v8f acc = {};
  for (int kt = 0; kt < DIM / 32; ++kt) {
    const float* xp = xrow + kt * 32;
    float4 x0 = *(const float4*)(xp);
    float4 x1 = *(const float4*)(xp + 4);
    float4 x2 = *(const float4*)(xp + 16);
    float4 x3 = *(const float4*)(xp + 20);
    v16bf av;
    av[0] = (bf16_t)x0.x;  av[1] = (bf16_t)x0.y;
    av[2] = (bf16_t)x0.z;  av[3] = (bf16_t)x0.w;
    av[4] = (bf16_t)x1.x;  av[5] = (bf16_t)x1.y;
    av[6] = (bf16_t)x1.z;  av[7] = (bf16_t)x1.w;
    av[8] = (bf16_t)x2.x;  av[9] = (bf16_t)x2.y;
    av[10] = (bf16_t)x2.z; av[11] = (bf16_t)x2.w;
    av[12] = (bf16_t)x3.x; av[13] = (bf16_t)x3.y;
    av[14] = (bf16_t)x3.z; av[15] = (bf16_t)x3.w;
    acc = wmma_bf16(av, fragB(wrow, lane, kt), acc);
  }
  const float Ah = -__expf(a_log[hcol]);
#pragma unroll
  for (int v = 0; v < 8; ++v) {
    int r = r0 + v + ((lane >> 4) << 3);
    int s = r & (SEQ - 1);
    float dt = softplus_f(acc[v] + dt_bias[s * HEADS + hcol]);
    dt_out[r * HEADS + hcol] = dt;
    dA_out[r * HEADS + hcol] = Ah * dt;
  }
}

// -------------------------------------------- per (b,chunk,head): Ydiag + states
__global__ void __launch_bounds__(128)
k_chunk(const float* __restrict__ X, const float* __restrict__ Bm,
        const float* __restrict__ Cm, const float* __restrict__ dtw,
        const float* __restrict__ dAw, float* __restrict__ Ttot,
        float* __restrict__ states, float* __restrict__ y) {
  __shared__ float  s_cs[BLOCK_L];
  __shared__ float  s_dt[BLOCK_L];
  __shared__ bf16_t sC[BLOCK_L][PAD];    // [l][n]   A for G
  __shared__ bf16_t sB[BLOCK_L][PAD];    // [l][n]   B rows for G (B^T K-major)
  __shared__ bf16_t sBt[DSTATE][PAD];    // [n][l]   B for states
  __shared__ bf16_t sXt[HEADDIM][PAD];   // [p][s]   (X*dt)^T, B for Y_diag
  __shared__ bf16_t sXdT[HEADDIM][PAD];  // [p][l]   (X*dt*decay)^T, A for states
  __shared__ bf16_t sM[BLOCK_L][PAD];    // [l][s]   masked (C B^T) ∘ L

  const int bid = blockIdx.x;
  const int h = bid & (HEADS - 1);
  const int c = (bid / HEADS) & (NCHUNK - 1);
  const int b = bid / (HEADS * NCHUNK);
  const int tid = threadIdx.x;
  const int lane = tid & 31;
  const int wv = tid >> 5;
  const int r0 = b * SEQ + c * BLOCK_L;  // flat (b,s) row base

  if (tid < BLOCK_L) {
    s_dt[tid] = dtw[(r0 + tid) * HEADS + h];
    s_cs[tid] = dAw[(r0 + tid) * HEADS + h];
  }
  __syncthreads();
  if (tid == 0) {
    float run = 0.f;
    for (int l = 0; l < BLOCK_L; ++l) { run += s_cs[l]; s_cs[l] = run; }
    Ttot[(b * HEADS + h) * NCHUNK + c] = run;
  }
  __syncthreads();
  const float T = s_cs[BLOCK_L - 1];

  // stage operands: vector global loads, scalar LDS stores (some transposed)
  for (int idx = tid; idx < (BLOCK_L * HEADDIM) / 4; idx += 128) {
    int l = idx >> 4;
    int q0 = (idx & 15) * 4;
    const size_t g = (size_t)(r0 + l) * DIM + h * DSTATE + q0;
    float4 xv = *(const float4*)&X[g];
    float4 bv = *(const float4*)&Bm[g];
    float4 cv = *(const float4*)&Cm[g];
    const float dtl = s_dt[l];
    const float dec = __expf(T - s_cs[l]);
#pragma unroll
    for (int j = 0; j < 4; ++j) {
      float xf = ((const float*)&xv)[j] * dtl;
      float bf = ((const float*)&bv)[j];
      float cf = ((const float*)&cv)[j];
      int q = q0 + j;
      sC[l][q] = (bf16_t)cf;
      sB[l][q] = (bf16_t)bf;
      sBt[q][l] = (bf16_t)bf;
      sXt[q][l] = (bf16_t)xf;
      sXdT[q][l] = (bf16_t)(xf * dec);
    }
  }
  __syncthreads();

  const int mrow = wv * 16 + (lane & 15);
  const int col = lane & 15;

  // --- G = C B^T, mask + decay -> sM ---
  for (int nt = 0; nt < 4; ++nt) {
    v8f acc = {};
#pragma unroll
    for (int kt = 0; kt < 2; ++kt)
      acc = wmma_bf16(fragA(&sC[mrow][0], lane, kt),
                      fragB(&sB[nt * 16 + col][0], lane, kt), acc);
#pragma unroll
    for (int v = 0; v < 8; ++v) {
      int i = wv * 16 + v + ((lane >> 4) << 3);
      int j = nt * 16 + col;
      float g = (j <= i) ? acc[v] * __expf(s_cs[i] - s_cs[j]) : 0.f;
      sM[i][j] = (bf16_t)g;
    }
  }
  __syncthreads();

  // --- Y_diag = M * (X dt) ---
  for (int nt = 0; nt < 4; ++nt) {
    v8f acc = {};
#pragma unroll
    for (int kt = 0; kt < 2; ++kt)
      acc = wmma_bf16(fragA(&sM[mrow][0], lane, kt),
                      fragB(&sXt[nt * 16 + col][0], lane, kt), acc);
#pragma unroll
    for (int v = 0; v < 8; ++v) {
      int i = wv * 16 + v + ((lane >> 4) << 3);
      int j = nt * 16 + col;
      y[(size_t)(r0 + i) * DIM + h * HEADDIM + j] = acc[v];
    }
  }

  // --- states[p][n] = sum_l Xdec[l][p] * Bc[l][n] ---
  float* st = states + (((size_t)b * NCHUNK + c) * HEADS + h) * (HEADDIM * DSTATE);
  for (int nt = 0; nt < 4; ++nt) {
    v8f acc = {};
#pragma unroll
    for (int kt = 0; kt < 2; ++kt)
      acc = wmma_bf16(fragA(&sXdT[mrow][0], lane, kt),
                      fragB(&sBt[nt * 16 + col][0], lane, kt), acc);
#pragma unroll
    for (int v = 0; v < 8; ++v) {
      int p = wv * 16 + v + ((lane >> 4) << 3);
      int n = nt * 16 + col;
      st[p * DSTATE + n] = acc[v];
    }
  }
}

// ------------------------------------------------ inter-chunk scan (per b,h)
__global__ void __launch_bounds__(256)
k_scan(const float* __restrict__ states, const float* __restrict__ Ttot,
       float* __restrict__ Pst, float* __restrict__ fstate) {
  const int b = blockIdx.x >> 4;
  const int h = blockIdx.x & 15;
  const int i0 = threadIdx.x * 16;
  float carry[16];
#pragma unroll
  for (int j = 0; j < 16; ++j) carry[j] = 0.f;
  for (int c = 0; c < NCHUNK; ++c) {
    const size_t off =
        (((size_t)b * NCHUNK + c) * HEADS + h) * (HEADDIM * DSTATE) + i0;
    const float dec = __expf(Ttot[(b * HEADS + h) * NCHUNK + c]);
#pragma unroll
    for (int q = 0; q < 4; ++q) {
      float4 sv = *(const float4*)&states[off + q * 4];
      float4 pv;
      pv.x = carry[q * 4 + 0]; pv.y = carry[q * 4 + 1];
      pv.z = carry[q * 4 + 2]; pv.w = carry[q * 4 + 3];
      *(float4*)&Pst[off + q * 4] = pv;
      carry[q * 4 + 0] = dec * pv.x + sv.x;
      carry[q * 4 + 1] = dec * pv.y + sv.y;
      carry[q * 4 + 2] = dec * pv.z + sv.z;
      carry[q * 4 + 3] = dec * pv.w + sv.w;
    }
  }
  const size_t fo = ((size_t)b * HEADS + h) * (HEADDIM * DSTATE) + i0;
#pragma unroll
  for (int j = 0; j < 16; ++j) fstate[fo + j] = carry[j];
}

// ------------------------------------------- Y_off += exp(cs) * (C * P^T)
__global__ void __launch_bounds__(128)
k_yoff(const float* __restrict__ Cm, const float* __restrict__ dAw,
       const float* __restrict__ Pst, float* __restrict__ y) {
  __shared__ float  s_cs[BLOCK_L];
  __shared__ bf16_t sC[BLOCK_L][PAD];   // [l][n]
  __shared__ bf16_t sP[HEADDIM][PAD];   // [p][n]  (K-major for B-frag)

  const int bid = blockIdx.x;
  const int h = bid & (HEADS - 1);
  const int c = (bid / HEADS) & (NCHUNK - 1);
  const int b = bid / (HEADS * NCHUNK);
  const int tid = threadIdx.x;
  const int lane = tid & 31;
  const int wv = tid >> 5;
  const int r0 = b * SEQ + c * BLOCK_L;

  if (tid < BLOCK_L) s_cs[tid] = dAw[(r0 + tid) * HEADS + h];
  __syncthreads();
  if (tid == 0) {
    float run = 0.f;
    for (int l = 0; l < BLOCK_L; ++l) { run += s_cs[l]; s_cs[l] = run; }
  }
  __syncthreads();

  const float* P =
      Pst + (((size_t)b * NCHUNK + c) * HEADS + h) * (HEADDIM * DSTATE);
  for (int idx = tid; idx < (BLOCK_L * DSTATE) / 4; idx += 128) {
    int l = idx >> 4;
    int q0 = (idx & 15) * 4;
    float4 cv = *(const float4*)&Cm[(size_t)(r0 + l) * DIM + h * DSTATE + q0];
    float4 pv = *(const float4*)&P[l * DSTATE + q0];
#pragma unroll
    for (int j = 0; j < 4; ++j) {
      sC[l][q0 + j] = (bf16_t)((const float*)&cv)[j];
      sP[l][q0 + j] = (bf16_t)((const float*)&pv)[j];
    }
  }
  __syncthreads();

  const int mrow = wv * 16 + (lane & 15);
  const int col = lane & 15;
  for (int nt = 0; nt < 4; ++nt) {
    v8f acc = {};
#pragma unroll
    for (int kt = 0; kt < 2; ++kt)
      acc = wmma_bf16(fragA(&sC[mrow][0], lane, kt),
                      fragB(&sP[nt * 16 + col][0], lane, kt), acc);
#pragma unroll
    for (int v = 0; v < 8; ++v) {
      int i = wv * 16 + v + ((lane >> 4) << 3);
      int j = nt * 16 + col;
      y[(size_t)(r0 + i) * DIM + h * HEADDIM + j] += __expf(s_cs[i]) * acc[v];
    }
  }
}

// ------------------------------------------------------------------- LayerNorm
__global__ void __launch_bounds__(256)
k_ln(const float* __restrict__ y, const float* __restrict__ gamma,
     const float* __restrict__ beta, bf16_t* __restrict__ yn) {
  __shared__ float red[256];
  const int r = blockIdx.x;
  const int tid = threadIdx.x;
  float v[4];
  float sum = 0.f;
#pragma unroll
  for (int j = 0; j < 4; ++j) {
    v[j] = y[(size_t)r * DIM + tid + j * 256];
    sum += v[j];
  }
  red[tid] = sum;
  __syncthreads();
  for (int s = 128; s > 0; s >>= 1) {
    if (tid < s) red[tid] += red[tid + s];
    __syncthreads();
  }
  const float mu = red[0] * (1.f / DIM);
  __syncthreads();
  float sq = 0.f;
#pragma unroll
  for (int j = 0; j < 4; ++j) { float d = v[j] - mu; sq += d * d; }
  red[tid] = sq;
  __syncthreads();
  for (int s = 128; s > 0; s >>= 1) {
    if (tid < s) red[tid] += red[tid + s];
    __syncthreads();
  }
  const float rstd = rsqrtf(red[0] * (1.f / DIM) + LN_EPS);
#pragma unroll
  for (int j = 0; j < 4; ++j) {
    int d = tid + j * 256;
    yn[(size_t)r * DIM + d] = (bf16_t)((v[j] - mu) * rstd * gamma[d] + beta[d]);
  }
}

// ------------- out = yn(bf16) @ Wout (WoutT is [n][k]); async double-buffered
__global__ void __launch_bounds__(128)
k_gemm(const bf16_t* __restrict__ A, const bf16_t* __restrict__ BwT,
       float* __restrict__ out) {
  __shared__ bf16_t sA[2][64][PAD];   // [buf][m][k]
  __shared__ bf16_t sW[2][64][PAD];   // [buf][n][k]
  const int n0 = blockIdx.x * 64;
  const int m0 = blockIdx.y * 64;
  const int tid = threadIdx.x;
  const int lane = tid & 31;
  const int wv = tid >> 5;
  v8f acc[4];
#pragma unroll
  for (int nt = 0; nt < 4; ++nt) acc[nt] = (v8f){};

  auto stage = [&](int buf, int k0) {
    for (int idx = tid; idx < 512; idx += 128) {
      int i = idx >> 3;
      int kc = (idx & 7) * 8;
      copy16_g2l(&sA[buf][i][kc], &A[(size_t)(m0 + i) * DIM + k0 + kc]);
      copy16_g2l(&sW[buf][i][kc], &BwT[(size_t)(n0 + i) * DIM + k0 + kc]);
    }
  };

  stage(0, 0);
  wait_async0();
  __syncthreads();
  for (int k0 = 0; k0 < DIM; k0 += 64) {
    const int cur = (k0 >> 6) & 1;
    if (k0 + 64 < DIM) stage(cur ^ 1, k0 + 64);  // prefetch next tile (async)
#pragma unroll
    for (int nt = 0; nt < 4; ++nt) {
#pragma unroll
      for (int kt = 0; kt < 2; ++kt)
        acc[nt] = wmma_bf16(fragA(&sA[cur][wv * 16 + (lane & 15)][0], lane, kt),
                            fragB(&sW[cur][nt * 16 + (lane & 15)][0], lane, kt),
                            acc[nt]);
    }
    wait_async0();
    __syncthreads();
  }
#pragma unroll
  for (int nt = 0; nt < 4; ++nt) {
#pragma unroll
    for (int v = 0; v < 8; ++v) {
      int i = m0 + wv * 16 + v + ((lane >> 4) << 3);
      int j = n0 + nt * 16 + (lane & 15);
      out[(size_t)i * DIM + j] = acc[nt][v];
    }
  }
}

// ----------------------------------------------------------------------- launch
extern "C" void kernel_launch(void* const* d_in, const int* in_sizes, int n_in,
                              void* d_out, int out_size, void* d_ws, size_t ws_size,
                              hipStream_t stream) {
  (void)in_sizes; (void)n_in; (void)out_size; (void)ws_size;
  const float* X        = (const float*)d_in[0];
  const float* B        = (const float*)d_in[1];
  const float* C        = (const float*)d_in[2];
  const float* Wdt      = (const float*)d_in[3];
  const float* dt_bias  = (const float*)d_in[4];
  const float* a_log    = (const float*)d_in[5];
  const float* W_out    = (const float*)d_in[6];
  const float* ln_scale = (const float*)d_in[7];
  const float* ln_bias  = (const float*)d_in[8];
  // d_in[9] = block_len (compile-time constant 64 here)

  float* out    = (float*)d_out;                    // (b,s,dim) f32
  float* fstate = out + (size_t)BATCH * SEQ * DIM;  // (b,h,p,n) f32

  char* ws = (char*)d_ws;
  size_t off = 0;
  auto alloc = [&](size_t bytes) -> char* {
    char* p = ws + off;
    off += (bytes + 255) & ~(size_t)255;
    return p;
  };
  const size_t SSTATE = (size_t)BATCH * NCHUNK * HEADS * HEADDIM * DSTATE;
  float*  dtw    = (float*)alloc(sizeof(float) * BATCH * SEQ * HEADS);
  float*  dAw    = (float*)alloc(sizeof(float) * BATCH * SEQ * HEADS);
  float*  Ttot   = (float*)alloc(sizeof(float) * BATCH * HEADS * NCHUNK);
  float*  states = (float*)alloc(sizeof(float) * SSTATE);
  float*  Pst    = (float*)alloc(sizeof(float) * SSTATE);
  float*  yb     = (float*)alloc(sizeof(float) * BATCH * SEQ * DIM);
  bf16_t* ynb    = (bf16_t*)alloc(sizeof(bf16_t) * BATCH * SEQ * DIM);
  bf16_t* WbT    = (bf16_t*)alloc(sizeof(bf16_t) * DIM * DIM);
  bf16_t* WdtT   = (bf16_t*)alloc(sizeof(bf16_t) * DIM * HEADS);

  k_prep<<<(DIM * DIM + 255) / 256, 256, 0, stream>>>(W_out, Wdt, WbT, WdtT);
  k_dt<<<BATCH * SEQ / 16, 32, 0, stream>>>(X, WdtT, dt_bias, a_log, dtw, dAw);
  k_chunk<<<BATCH * NCHUNK * HEADS, 128, 0, stream>>>(X, B, C, dtw, dAw, Ttot,
                                                      states, yb);
  k_scan<<<BATCH * HEADS, 256, 0, stream>>>(states, Ttot, Pst, fstate);
  k_yoff<<<BATCH * NCHUNK * HEADS, 128, 0, stream>>>(C, dAw, Pst, yb);
  k_ln<<<BATCH * SEQ, 256, 0, stream>>>(yb, ln_scale, ln_bias, ynb);
  dim3 g(DIM / 64, BATCH * SEQ / 64);
  k_gemm<<<g, 128, 0, stream>>>(ynb, WbT, out);
}
